// PointTransformerLayer_60825326846158
// MI455X (gfx1250) — matrix operand order repaired and checked by reading, not compiled, not run on hardware
//
#include <hip/hip_runtime.h>

typedef __attribute__((ext_vector_type(16))) _Float16 v16h;
typedef __attribute__((ext_vector_type(8)))  _Float16 v8h;
typedef __attribute__((ext_vector_type(8)))  float    v8f;
typedef __attribute__((ext_vector_type(4)))  unsigned int u32x4;
typedef __attribute__((ext_vector_type(8)))  int      i32x8;
typedef __attribute__((ext_vector_type(4)))  int      i32x4;

#define B_   4
#define N_   8192
#define CIN  64
#define CO   64
#define KNN_ 16
#define LDT  72          // f16 row stride: 144B = 36 banks, 16B-aligned rows
#define WROW 72          // padded row stride for staged weights (matches TDM pad)
#define WMAT (64 * WROW) // halves per staged 64x64 weight matrix

// ---------------------------------------------------------------- helpers
__device__ __forceinline__ v8f wmma_f16(v16h a, v16h b, v8f c) {
  return __builtin_amdgcn_wmma_f32_16x16x32_f16(
      /*neg_a=*/false, a, /*neg_b=*/false, b,
      /*c_mod=*/(short)0, c, /*reuse_a=*/false, /*reuse_b=*/false);
}

__device__ __forceinline__ v16h zero16h() {
  v16h z;
#pragma unroll
  for (int i = 0; i < 16; ++i) z[i] = (_Float16)0.0f;
  return z;
}
__device__ __forceinline__ v8f zero8f() {
  v8f z;
#pragma unroll
  for (int i = 0; i < 8; ++i) z[i] = 0.0f;
  return z;
}
__device__ __forceinline__ v16h cat8(v8h lo, v8h hi) {
  return __builtin_shufflevector(lo, hi, 0, 1, 2, 3, 4, 5, 6, 7,
                                 8, 9, 10, 11, 12, 13, 14, 15);
}

// A operand (16-bit, 16xK): row m = lane&15; elems = 16 consecutive halves [base..base+15]
__device__ __forceinline__ v16h load_A_lds(const _Float16* t, int lane, int kblock) {
  const int m = lane & 15, h = lane >> 4;
  const _Float16* p = t + m * LDT + kblock * 32 + h * 16;
  return cat8(*(const v8h*)(p), *(const v8h*)(p + 8));
}

// B operand (16-bit, Kx16): col o = lane&15 (+obase); elems = 16 consecutive halves
__device__ __forceinline__ v16h load_B_row(const _Float16* w, int lane, int obase,
                                           int kblock, int ld) {
  const int o  = obase + (lane & 15);
  const int kb = kblock * 32 + ((lane >> 4) ? 16 : 0);
  const _Float16* p = w + o * ld + kb;
  return cat8(*(const v8h*)(p), *(const v8h*)(p + 8));
}

// ---------------------------------------------------------------- TDM staging
#if defined(__has_builtin)
#if __has_builtin(__builtin_amdgcn_tensor_load_to_lds)
#define HAVE_TDM 1
#endif
#endif

#ifdef HAVE_TDM
// Stage one 64x64 f16 matrix (4096 contiguous halves) into LDS with
// pad_interval=32 dwords (one 64-half row), pad_amount=4 dwords (8 halves)
// -> LDS row stride WROW=72 halves, matching load_B_row's conflict-free layout.
__device__ __forceinline__ void tdm_load_w(const _Float16* src, unsigned int lds_off) {
  const unsigned long long ga = (unsigned long long)src;
  u32x4 g0;
  g0[0] = 1u;                                    // count=1, user mode
  g0[1] = lds_off;                               // lds_addr (bytes)
  g0[2] = (unsigned int)(ga & 0xffffffffu);      // global_addr[31:0]
  g0[3] = (unsigned int)(ga >> 32) | (2u << 30); // global_addr[56:32] | type=2
  i32x8 g1;
  g1[0] = (int)((1u << 16)        // data_size = 2 bytes
              | (1u << 20)        // pad_enable
              | (4u << 22)        // pad_interval: 32 dwords
              | (3u << 25));      // pad_amount: 4 dwords
  g1[1] = (int)(4096u << 16);     // tensor_dim0 = 4096 (low 16 in [31:16])
  g1[2] = (int)(1u << 16);        // tensor_dim0 hi=0 | tensor_dim1 = 1
  g1[3] = (int)(4096u << 16);     // tensor_dim1 hi=0 | tile_dim0 = 4096
  g1[4] = 1;                      // tile_dim1 = 1, tile_dim2 = 0
  g1[5] = 4096;                   // tensor_dim0_stride low32
  g1[6] = (int)(4096u << 16);     // stride0 hi=0 | tensor_dim1_stride low16
  g1[7] = 0;
  const i32x4 z4 = {0, 0, 0, 0};
  const i32x8 z8 = {0, 0, 0, 0, 0, 0, 0, 0};
  __builtin_amdgcn_tensor_load_to_lds(g0, g1, z4, z4, z8, 0);
}
#endif

// ---------------------------------------------------------------- convert
// fhT: features transposed to (b, n, c) in f16 so GEMM B operands are
// 16 contiguous halves (two b128 loads) instead of 16 strided 16-bit loads.
__global__ __launch_bounds__(256) void convert_kernel(
    const float* __restrict__ features, const float* __restrict__ wlin,
    const float* __restrict__ wt1, const float* __restrict__ wt2,
    const float* __restrict__ wg1, const float* __restrict__ wg2,
    _Float16* __restrict__ fhT, _Float16* __restrict__ wlinh,
    _Float16* __restrict__ wt1h, _Float16* __restrict__ wt2h,
    _Float16* __restrict__ wg1h, _Float16* __restrict__ wg2h) {
  const int tid = blockIdx.x * 256 + threadIdx.x;
  const int FT = B_ * CIN * N_;
  if (tid < FT) {
    const int n = tid % N_;
    const int c = (tid / N_) % CIN;
    const int b = tid / (N_ * CIN);
    fhT[((size_t)b * N_ + n) * CIN + c] = (_Float16)features[tid];
  }
  if (tid < 192 * 64) wlinh[tid] = (_Float16)wlin[tid];
  if (tid < 64 * 4) {
    const int o = tid >> 2, c = tid & 3;
    wt1h[tid] = (_Float16)((c < 3) ? wt1[o * 3 + c] : 0.0f);
  }
  if (tid < 64 * 64) {
    wt2h[tid] = (_Float16)wt2[tid];
    wg1h[tid] = (_Float16)wg1[tid];
    wg2h[tid] = (_Float16)wg2[tid];
  }
}

// ---------------------------------------------------------------- KNN
__global__ __launch_bounds__(256) void knn_kernel(const float* __restrict__ coords,
                                                  int* __restrict__ idx) {
  __shared__ __align__(16) float sx[256], sy[256], sz[256];
  const int b = blockIdx.x / (N_ / 256);
  const int q = (blockIdx.x % (N_ / 256)) * 256 + threadIdx.x;
  const float* cx = coords + (size_t)(b * 3 + 0) * N_;
  const float* cy = coords + (size_t)(b * 3 + 1) * N_;
  const float* cz = coords + (size_t)(b * 3 + 2) * N_;
  const float qx = cx[q], qy = cy[q], qz = cz[q];

  const unsigned int lx = (unsigned int)(size_t)&sx[threadIdx.x];
  const unsigned int ly = (unsigned int)(size_t)&sy[threadIdx.x];
  const unsigned int lz = (unsigned int)(size_t)&sz[threadIdx.x];

  float bd[KNN_];
  int   bi[KNN_];
#pragma unroll
  for (int i = 0; i < KNN_; ++i) { bd[i] = 3.0e38f; bi[i] = 0; }
  float worst = 3.0e38f;
  int   worstp = 0;

  for (int base = 0; base < N_; base += 256) {
    __syncthreads();  // previous chunk fully consumed
    {
      const unsigned long long gx = (unsigned long long)(cx + base + threadIdx.x);
      const unsigned long long gy = (unsigned long long)(cy + base + threadIdx.x);
      const unsigned long long gz = (unsigned long long)(cz + base + threadIdx.x);
      asm volatile("global_load_async_to_lds_b32 %0, %1, off"
                   :: "v"(lx), "v"(gx) : "memory");
      asm volatile("global_load_async_to_lds_b32 %0, %1, off"
                   :: "v"(ly), "v"(gy) : "memory");
      asm volatile("global_load_async_to_lds_b32 %0, %1, off"
                   :: "v"(lz), "v"(gz) : "memory");
      asm volatile("s_wait_asynccnt 0x0" ::: "memory");
    }
    __syncthreads();
    for (int jj = 0; jj < 256; ++jj) {
      const float dx = qx - sx[jj], dy = qy - sy[jj], dz = qz - sz[jj];
      const float d = dx * dx + dy * dy + dz * dz;
      if (d < worst) {
        // predicated replace-max (static indices -> register-resident)
#pragma unroll
        for (int i = 0; i < KNN_; ++i)
          if (i == worstp) { bd[i] = d; bi[i] = base + jj; }
        worst = bd[0]; worstp = 0;
#pragma unroll
        for (int i = 1; i < KNN_; ++i)
          if (bd[i] > worst) { worst = bd[i]; worstp = i; }
      }
    }
  }
#pragma unroll
  for (int i = 0; i < KNN_; ++i)
    idx[((size_t)b * N_ + q) * KNN_ + i] = bi[i];
}

// ---------------------------------------------------------------- feat3 GEMM
// feat3[b, row, n] = sum_c wlin[row, c] * fhT[b, n, c]  (row in 0..191)
__global__ __launch_bounds__(256) void feat_kernel(const _Float16* __restrict__ fhT,
                                                   const _Float16* __restrict__ wlinh,
                                                   float* __restrict__ feat3) {
  const int wave = threadIdx.x >> 5;
  const int lane = threadIdx.x & 31;
  const int q = lane & 15, h = lane >> 4;

  const int cb = blockIdx.x & 63;            // 64 column blocks of 128
  const int rt = (blockIdx.x >> 6) % 12;     // 12 row tiles of 16
  const int b  = blockIdx.x / (64 * 12);
  const int ncol = cb * 128 + wave * 16 + q;
  const int row  = rt * 16 + q;

  v8f acc = zero8f();
#pragma unroll
  for (int kb = 0; kb < 2; ++kb) {
    const _Float16* pa = wlinh + row * 64 + kb * 32 + h * 16;
    const v16h a = cat8(*(const v8h*)pa, *(const v8h*)(pa + 8));
    // B(c, n) = fhT[b, n, c]: 16 contiguous halves at (n=ncol, c=kb*32+h*16)
    const _Float16* pb = fhT + ((size_t)b * N_ + ncol) * CIN + kb * 32 + h * 16;
    const v16h bb = cat8(*(const v8h*)pb, *(const v8h*)(pb + 8));
    acc = wmma_f16(a, bb, acc);
  }
#pragma unroll
  for (int r = 0; r < 8; ++r) {
    const int rrow = rt * 16 + (h ? r + 8 : r);
    feat3[((size_t)b * 192 + rrow) * N_ + ncol] = acc[r];
  }
}

// ---------------------------------------------------------------- attention
// One point per wave (4 waves / 128-thread block). M=16 neighbor rows.
__global__ __launch_bounds__(128) void attn_kernel(
    const float* __restrict__ coords, const float* __restrict__ feat3,
    const int* __restrict__ idx,
    const _Float16* __restrict__ wt1h, const _Float16* __restrict__ wt2h,
    const _Float16* __restrict__ wg1h, const _Float16* __restrict__ wg2h,
    float* __restrict__ out) {
  __shared__ __align__(16) _Float16 tbuf[4][16 * LDT];
  __shared__ __align__(16) _Float16 wbuf[3 * WMAT];   // wt2 | wg1 | wg2, row stride 72
  __shared__ int jbuf[4][KNN_];

  const int wave = threadIdx.x >> 5;
  const int lane = threadIdx.x & 31;
  const int q = lane & 15, h = lane >> 4;
  const int gid = blockIdx.x * 4 + wave;   // 0 .. B*N-1
  const int b = gid / N_;
  const int n = gid % N_;
  _Float16* t = &tbuf[wave][0];
  const _Float16* sw2 = &wbuf[0];
  const _Float16* sg1 = &wbuf[WMAT];
  const _Float16* sg2 = &wbuf[2 * WMAT];

  // ---- stage the three 64x64 weight matrices into LDS (TDM if available) ----
#ifdef HAVE_TDM
  if (wave == 0) {
    const unsigned int wb = (unsigned int)(size_t)&wbuf[0];
    tdm_load_w(wt2h, wb);
    tdm_load_w(wg1h, wb + WMAT * 2u);
    tdm_load_w(wg2h, wb + 2u * WMAT * 2u);
    __builtin_amdgcn_s_wait_tensorcnt(0);
  }
  __syncthreads();
#else
  for (int i = threadIdx.x; i < 3 * 4096; i += 128) {
    const int mat = i >> 12, e = i & 4095;
    const _Float16* src = (mat == 0) ? wt2h : ((mat == 1) ? wg1h : wg2h);
    wbuf[mat * WMAT + (e >> 6) * WROW + (e & 63)] = src[e];
  }
  __syncthreads();
#endif

  // neighbor + relative coords for row m = q (both halves compute same row set)
  const int jm = idx[((size_t)b * N_ + n) * KNN_ + q];
  if (h == 0) jbuf[wave][q] = jm;
  const float rx = coords[(size_t)(b * 3 + 0) * N_ + n] - coords[(size_t)(b * 3 + 0) * N_ + jm];
  const float ry = coords[(size_t)(b * 3 + 1) * N_ + n] - coords[(size_t)(b * 3 + 1) * N_ + jm];
  const float rz = coords[(size_t)(b * 3 + 2) * N_ + n] - coords[(size_t)(b * 3 + 2) * N_ + jm];

  // ---- theta1: A = rel (16x3 padded to 16x32), B = wt1 (padded ld=4) ----
  v16h a1 = zero16h();
  if (h == 0) { a1[0] = (_Float16)rx; a1[1] = (_Float16)ry; a1[2] = (_Float16)rz; }
  v8f t1[4];
#pragma unroll
  for (int tn = 0; tn < 4; ++tn) {
    v16h bb = zero16h();
    const int o = tn * 16 + q;
    if (h == 0) { bb[0] = wt1h[o * 4 + 0]; bb[1] = wt1h[o * 4 + 1]; bb[2] = wt1h[o * 4 + 2]; }
    t1[tn] = wmma_f16(a1, bb, zero8f());
  }
  // store t1 (D layout -> row-major LDS f16)
#pragma unroll
  for (int tn = 0; tn < 4; ++tn)
#pragma unroll
    for (int r = 0; r < 8; ++r)
      t[(h ? r + 8 : r) * LDT + tn * 16 + q] = (_Float16)t1[tn][r];

  // ---- theta2 (relu) -> delta kept in fp32 registers ----
  v8f dlt[4];
  {
    const v16h a0 = load_A_lds(t, lane, 0);
    const v16h ak = load_A_lds(t, lane, 1);
#pragma unroll
    for (int tn = 0; tn < 4; ++tn) {
      v8f acc = zero8f();
      acc = wmma_f16(a0, load_B_row(sw2, lane, tn * 16, 0, WROW), acc);
      acc = wmma_f16(ak, load_B_row(sw2, lane, tn * 16, 1, WROW), acc);
#pragma unroll
      for (int r = 0; r < 8; ++r) acc[r] = fmaxf(acc[r], 0.0f);
      dlt[tn] = acc;
    }
  }

  // ---- inputs_gamma = phi - psi_g + delta  -> LDS (as A for gamma1) ----
#pragma unroll
  for (int tn = 0; tn < 4; ++tn) {
    const int c = tn * 16 + q;
    const float phic = feat3[((size_t)b * 192 + c) * N_ + n];
#pragma unroll
    for (int r = 0; r < 8; ++r) {
      const int mm = h ? r + 8 : r;
      const int j = jbuf[wave][mm];
      const float psij = feat3[((size_t)b * 192 + 64 + c) * N_ + j];
      t[mm * LDT + c] = (_Float16)(phic - psij + dlt[tn][r]);
    }
  }

  // ---- gamma1 (no relu) -> LDS ----
  v8f g1t[4];
  {
    const v16h a0 = load_A_lds(t, lane, 0);
    const v16h ak = load_A_lds(t, lane, 1);
#pragma unroll
    for (int tn = 0; tn < 4; ++tn) {
      v8f acc = zero8f();
      acc = wmma_f16(a0, load_B_row(sg1, lane, tn * 16, 0, WROW), acc);
      acc = wmma_f16(ak, load_B_row(sg1, lane, tn * 16, 1, WROW), acc);
      g1t[tn] = acc;
    }
  }
#pragma unroll
  for (int tn = 0; tn < 4; ++tn)
#pragma unroll
    for (int r = 0; r < 8; ++r)
      t[(h ? r + 8 : r) * LDT + tn * 16 + q] = (_Float16)g1t[tn][r];

  // ---- gamma2 (relu) -> g in registers ----
  v8f g[4];
  {
    const v16h a0 = load_A_lds(t, lane, 0);
    const v16h ak = load_A_lds(t, lane, 1);
#pragma unroll
    for (int tn = 0; tn < 4; ++tn) {
      v8f acc = zero8f();
      acc = wmma_f16(a0, load_B_row(sg2, lane, tn * 16, 0, WROW), acc);
      acc = wmma_f16(ak, load_B_row(sg2, lane, tn * 16, 1, WROW), acc);
#pragma unroll
      for (int r = 0; r < 8; ++r) acc[r] = fmaxf(acc[r], 0.0f);
      g[tn] = acc;
    }
  }

  // ---- softmax over the 16 neighbors (rows split across lane halves) ----
#pragma unroll
  for (int tn = 0; tn < 4; ++tn) {
    const int o = tn * 16 + q;
    float mx = g[tn][0];
#pragma unroll
    for (int r = 1; r < 8; ++r) mx = fmaxf(mx, g[tn][r]);
    mx = fmaxf(mx, __shfl_xor(mx, 16));
    float e[8], s = 0.0f;
#pragma unroll
    for (int r = 0; r < 8; ++r) { e[r] = __expf(g[tn][r] - mx); s += e[r]; }
    s += __shfl_xor(s, 16);
    const float inv = 1.0f / s;
    float part = 0.0f;
#pragma unroll
    for (int r = 0; r < 8; ++r) {
      const int mm = h ? r + 8 : r;
      const int j = jbuf[wave][mm];
      const float ag = feat3[((size_t)b * 192 + 128 + o) * N_ + j] + dlt[tn][r];
      part += e[r] * inv * ag;
    }
    part += __shfl_xor(part, 16);
    if (h == 0) out[((size_t)b * CO + o) * N_ + n] = part;
  }
}

// ---------------------------------------------------------------- launch
extern "C" void kernel_launch(void* const* d_in, const int* in_sizes, int n_in,
                              void* d_out, int out_size, void* d_ws, size_t ws_size,
                              hipStream_t stream) {
  const float* features = (const float*)d_in[0];
  const float* coords   = (const float*)d_in[1];
  const float* w_lin    = (const float*)d_in[2];
  const float* w_theta1 = (const float*)d_in[3];
  const float* w_theta2 = (const float*)d_in[4];
  const float* w_gamma1 = (const float*)d_in[5];
  const float* w_gamma2 = (const float*)d_in[6];
  float* out = (float*)d_out;

  char* ws = (char*)d_ws;
  int*      d_idx  = (int*)(ws);                                    // 2 MiB
  float*    d_f3   = (float*)(ws + (size_t)(2)  * 1024 * 1024);     // 24 MiB
  _Float16* d_fhT  = (_Float16*)(ws + (size_t)(26) * 1024 * 1024);  // 4 MiB
  _Float16* d_wlh  = (_Float16*)(ws + (size_t)(30) * 1024 * 1024);  // 24 KiB
  _Float16* d_wt1h = (_Float16*)(ws + (size_t)(30) * 1024 * 1024 + 32768);
  _Float16* d_wt2h = d_wt1h + 512;
  _Float16* d_wg1h = d_wt2h + 4096;
  _Float16* d_wg2h = d_wg1h + 4096;

  // 1) fp32 -> f16 conversions (+ feature transpose to (b, n, c))
  {
    const int total = B_ * CIN * N_;
    convert_kernel<<<(total + 255) / 256, 256, 0, stream>>>(
        features, w_lin, w_theta1, w_theta2, w_gamma1, w_gamma2,
        d_fhT, d_wlh, d_wt1h, d_wt2h, d_wg1h, d_wg2h);
  }
  // 2) KNN: one query per thread, async LDS staging
  knn_kernel<<<B_ * (N_ / 256), 256, 0, stream>>>(coords, d_idx);
  // 3) feat3 = w_lin @ features (WMMA GEMM)
  feat_kernel<<<B_ * 12 * 64, 256, 0, stream>>>(d_fhT, d_wlh, d_f3);
  // 4) attention: one point per wave, TDM-staged weights
  attn_kernel<<<(B_ * N_) / 4, 128, 0, stream>>>(
      coords, d_f3, d_idx, d_wt1h, d_wt2h, d_wg1h, d_wg2h, out);
}
